// PSA_1494648619602
// MI455X (gfx1250) — matrix operand order repaired and checked by reading, not compiled, not run on hardware
//
#include <hip/hip_runtime.h>
#include <hip/hip_bf16.h>
#include <cstddef>
#include <cstdint>

typedef __attribute__((ext_vector_type(16))) __bf16 v16bf;
typedef __attribute__((ext_vector_type(8)))  __bf16 v8bf;
typedef __attribute__((ext_vector_type(8)))  float  v8f;

#define NPIX 2304   // 48*48
#define BATCH 4

// ---------------------------------------------------------------------------
// WMMA fragment loaders (LDS rows hold contiguous K)
// A matrix 16x32 bf16 (ISA 7.12.2): lane<16 -> K {0..7,16..23}; lane>=16 -> {8..15,24..31}
__device__ __forceinline__ v16bf fragA(const __bf16* row, int lane) {
    int o = (lane >= 16) ? 8 : 0;
    v8bf lo = *(const v8bf*)(row + o);
    v8bf hi = *(const v8bf*)(row + 16 + o);
    v16bf r;
#pragma unroll
    for (int i = 0; i < 8; ++i) { r[i] = lo[i]; r[i + 8] = hi[i]; }
    return r;
}
// B matrix 32x16 bf16: lane<16 -> K {0..7,16..23}; lane>=16 -> {8..15,24..31}
__device__ __forceinline__ v16bf fragB(const __bf16* row, int lane) {
    int o = (lane >= 16) ? 8 : 0;
    v8bf lo = *(const v8bf*)(row + o);
    v8bf hi = *(const v8bf*)(row + 16 + o);
    v16bf r;
#pragma unroll
    for (int i = 0; i < 8; ++i) { r[i] = lo[i]; r[i + 8] = hi[i]; }
    return r;
}

__device__ __forceinline__ float silu(float v) { return v / (1.0f + __expf(-v)); }

// CDNA5 async DMA: 16B global -> LDS per lane, tracked by ASYNCcnt.
__device__ __forceinline__ void async_copy16(const void* lds_dst, uint64_t sbase,
                                             unsigned voff_bytes) {
    unsigned l = (unsigned)(uintptr_t)lds_dst;   // low 32 bits of LDS flat addr = LDS offset
    asm volatile("global_load_async_to_lds_b128 %0, %1, %2"
                 :: "v"(l), "v"(voff_bytes), "s"(sbase) : "memory");
}
__device__ __forceinline__ void async_wait0() {
    asm volatile("s_wait_asynccnt 0x0" ::: "memory");
}

// ---------------------------------------------------------------------------
__global__ __launch_bounds__(256) void f32_to_bf16_kernel(const float* __restrict__ in,
                                                          __bf16* __restrict__ out, size_t n) {
    for (size_t i = blockIdx.x * (size_t)blockDim.x + threadIdx.x; i < n;
         i += (size_t)gridDim.x * blockDim.x)
        out[i] = (__bf16)in[i];
}

// ---------------------------------------------------------------------------
__device__ __forceinline__ const __bf16* xrow(const __bf16* X0, int x0_ch, int x0_off,
                                              const __bf16* X1, int x1_ch, int x1_off,
                                              int ksplit, int b, int k, int kB) {
    if (k < ksplit) return X0 + (size_t)(b * x0_ch + x0_off + k + kB) * NPIX;
    return X1 + (size_t)(b * x1_ch + x1_off + (k - ksplit) + kB) * NPIX;
}

// Generic bf16 GEMM for 1x1 convs: Y[o,n] = sum_k W[o,k] * X[k,n], per batch b=blockIdx.z.
// Double-buffered LDS, one barrier per k-step; global prefetch 2 tiles ahead.
__global__ __launch_bounds__(256) void gemm_bf16_kernel(
    const __bf16* __restrict__ X0, int x0_ch, int x0_off,
    const __bf16* __restrict__ X1, int x1_ch, int x1_off, int ksplit,
    const float* __restrict__ Wt, int K, int M,
    const float* __restrict__ sc, const float* __restrict__ bi,
    const float* __restrict__ resid, int res_ch, int res_off,
    float* __restrict__ outF, int of_ch,
    __bf16* __restrict__ outB, int ob_ch,
    int act)
{
    __shared__ __align__(16) __bf16 As[2][128][32];   // [m][k]
    __shared__ __align__(16) __bf16 Bs[2][128][32];   // [n][k]
    const int b     = blockIdx.z;
    const int mBase = blockIdx.x * 128;
    const int nBase = blockIdx.y * 128;
    const int tid   = threadIdx.x;
    const int lane  = tid & 31;
    const int lh    = lane & 15;
    const int w     = tid >> 5;
    const int wm    = w >> 1;       // 0..3 -> 32 rows each
    const int wn    = w & 1;        // 0..1 -> 64 cols each

    const int rA = tid >> 1, kA = (tid & 1) * 16;   // A staging: 2 thr/row, 16 k each
    const int kB = tid >> 3, nB = (tid & 7) * 16;   // B staging: 8 thr/k-row, 16 n each

    v8f acc[2][4];
#pragma unroll
    for (int mi = 0; mi < 2; ++mi)
#pragma unroll
        for (int ni = 0; ni < 4; ++ni)
#pragma unroll
            for (int e = 0; e < 8; ++e) acc[mi][ni][e] = 0.0f;

    float  aw[16];
    __bf16 bx[16];
    { // prologue: fetch tile kt=0
        const float* srcA = Wt + (size_t)(mBase + rA) * K + kA;
#pragma unroll
        for (int j = 0; j < 16; ++j) aw[j] = srcA[j];
        const __bf16* srcB =
            xrow(X0, x0_ch, x0_off, X1, x1_ch, x1_off, ksplit, b, 0, kB) + nBase + nB;
#pragma unroll
        for (int j = 0; j < 16; ++j) bx[j] = srcB[j];
#pragma unroll
        for (int j = 0; j < 16; ++j) As[0][rA][kA + j] = (__bf16)aw[j];
#pragma unroll
        for (int j = 0; j < 16; ++j) Bs[0][nB + j][kB] = bx[j];
    }
    __syncthreads();

    int buf = 0;
    for (int kt = 0; kt < K; kt += 32) {
        const bool hn = (kt + 32) < K;
        if (hn) { // prefetch next tile into registers while this tile computes
            const int kn = kt + 32;
            const float* srcA = Wt + (size_t)(mBase + rA) * K + kn + kA;
            const __bf16* srcB =
                xrow(X0, x0_ch, x0_off, X1, x1_ch, x1_off, ksplit, b, kn, kB) + nBase + nB;
            if (kt + 64 < K) { // gfx1250 global_prefetch two tiles ahead
                __builtin_prefetch(srcA + 32, 0, 3);
                __builtin_prefetch(
                    xrow(X0, x0_ch, x0_off, X1, x1_ch, x1_off, ksplit, b, kt + 64, kB) +
                        nBase + nB, 0, 3);
            }
#pragma unroll
            for (int j = 0; j < 16; ++j) aw[j] = srcA[j];
#pragma unroll
            for (int j = 0; j < 16; ++j) bx[j] = srcB[j];
        }

        v16bf af[2];
#pragma unroll
        for (int mi = 0; mi < 2; ++mi)
            af[mi] = fragA(&As[buf][wm * 32 + mi * 16 + lh][0], lane);
#pragma unroll
        for (int ni = 0; ni < 4; ++ni) {
            v16bf bfr = fragB(&Bs[buf][wn * 64 + ni * 16 + lh][0], lane);
#pragma unroll
            for (int mi = 0; mi < 2; ++mi)
                acc[mi][ni] = __builtin_amdgcn_wmma_f32_16x16x32_bf16(
                    false, af[mi], false, bfr, (short)0, acc[mi][ni], false, false);
        }

        if (hn) { // fill the other buffer
            const int nb = buf ^ 1;
#pragma unroll
            for (int j = 0; j < 16; ++j) As[nb][rA][kA + j] = (__bf16)aw[j];
#pragma unroll
            for (int j = 0; j < 16; ++j) Bs[nb][nB + j][kB] = bx[j];
        }
        __syncthreads();
        buf ^= 1;
    }

    const int rowAdd = (lane >= 16) ? 8 : 0;
#pragma unroll
    for (int mi = 0; mi < 2; ++mi)
#pragma unroll
        for (int ni = 0; ni < 4; ++ni) {
            int nc = nBase + wn * 64 + ni * 16 + lh;
#pragma unroll
            for (int r = 0; r < 8; ++r) {
                int o = mBase + wm * 32 + mi * 16 + r + rowAdd;
                float v = acc[mi][ni][r] * sc[o] + bi[o];
                if (resid) v += resid[(size_t)(b * res_ch + res_off + o) * NPIX + nc];
                if (act) v = silu(v);
                if (outF) outF[(size_t)(b * of_ch + o) * NPIX + nc] = v;
                if (outB) outB[(size_t)(b * ob_ch + o) * NPIX + nc] = (__bf16)v;
            }
        }
}

// ---------------------------------------------------------------------------
// Flash attention. qkv layout (B, 512, NPIX): per head h, rows h*128+[0..31]=q,
// [32..63]=k, [64..127]=v. grid = (NPIX/128, B*4). Block 256 = 8 waves, each
// wave owns 16 queries. j processed in tiles of 64 with online softmax.
// V tile is staged with CDNA5 async global->LDS DMA (ASYNCcnt).
__global__ __launch_bounds__(256) void flash_attn_kernel(const __bf16* __restrict__ qkv,
                                                         float* __restrict__ outp,
                                                         float scale)
{
    __shared__ __align__(16) __bf16 Qs[128][32];       // [i][d]
    __shared__ __align__(16) __bf16 Ks[64][32];        // [j][d]
    __shared__ __align__(16) __bf16 Vs[64][64];        // [d][j]
    __shared__ __align__(16) __bf16 Ps[8][16][64];     // per-wave P: [i][j]
    __shared__ float fac_s[8][16];
    __shared__ float lsum_s[8][16];

    const int bh = blockIdx.y, b = bh >> 2, h = bh & 3;
    const int iBase = blockIdx.x * 128;
    const int tid = threadIdx.x, lane = tid & 31, lh = lane & 15, w = tid >> 5;
    const int rowAdd = (lane >= 16) ? 8 : 0;

    const __bf16* qp = qkv + (size_t)(b * 512 + h * 128) * NPIX;
    const __bf16* kp = qp + (size_t)32 * NPIX;
    const __bf16* vp = qp + (size_t)64 * NPIX;
    const uint64_t vbase = (uint64_t)(uintptr_t)vp;

    { // stage Q transposed once: Qs[i][d]
        int d = tid >> 3, i0 = (tid & 7) * 16;
#pragma unroll
        for (int j = 0; j < 16; ++j) Qs[i0 + j][d] = qp[(size_t)d * NPIX + iBase + i0 + j];
    }
    __syncthreads();
    v16bf aq = fragA(&Qs[w * 16 + lh][0], lane);

    v8f Oacc[4];
#pragma unroll
    for (int t = 0; t < 4; ++t)
#pragma unroll
        for (int e = 0; e < 8; ++e) Oacc[t][e] = 0.0f;
    float m[8], l[8];
#pragma unroll
    for (int r = 0; r < 8; ++r) { m[r] = -1e30f; l[r] = 0.0f; }

    for (int jt = 0; jt < NPIX; jt += 64) {
        __syncthreads();
        { // V tile: async DMA straight into LDS (layout matches memory)
#pragma unroll
            for (int i = 0; i < 2; ++i) {
                int g = tid + i * 256;               // 512 x 16B groups
                int d = g >> 3, j0 = (g & 7) * 8;
                async_copy16(&Vs[d][j0], vbase,
                             (unsigned)(((size_t)d * NPIX + jt + j0) * 2));
            }
        }
        { // K tile transposed via VGPRs: Ks[j][d]
            int d = tid >> 3, j0 = (tid & 7) * 8;
#pragma unroll
            for (int jj = 0; jj < 8; ++jj) Ks[j0 + jj][d] = kp[(size_t)d * NPIX + jt + j0 + jj];
        }
        async_wait0();
        __syncthreads();

        // S = scale * Q^T K  (four 16x16 j-subtiles, single K=32 WMMA each)
        v8f s[4];
#pragma unroll
        for (int t = 0; t < 4; ++t) {
            v16bf bk = fragB(&Ks[t * 16 + lh][0], lane);
            v8f z;
#pragma unroll
            for (int e = 0; e < 8; ++e) z[e] = 0.0f;
            s[t] = __builtin_amdgcn_wmma_f32_16x16x32_bf16(false, aq, false, bk, (short)0, z, false, false);
#pragma unroll
            for (int e = 0; e < 8; ++e) s[t][e] *= scale;
        }

        // online softmax, rows = queries (VGPR idx + lane-half), cols = lanes 0..15
        float fac[8];
#pragma unroll
        for (int r = 0; r < 8; ++r) {
            float mx = s[0][r];
#pragma unroll
            for (int t = 1; t < 4; ++t) mx = fmaxf(mx, s[t][r]);
            mx = fmaxf(mx, __shfl_xor(mx, 1, 32));
            mx = fmaxf(mx, __shfl_xor(mx, 2, 32));
            mx = fmaxf(mx, __shfl_xor(mx, 4, 32));
            mx = fmaxf(mx, __shfl_xor(mx, 8, 32));
            float mn = fmaxf(m[r], mx);
            fac[r] = __expf(m[r] - mn);
            float sum = 0.0f;
#pragma unroll
            for (int t = 0; t < 4; ++t) {
                float p = __expf(s[t][r] - mn);
                s[t][r] = p;
                sum += p;
            }
            sum += __shfl_xor(sum, 1, 32);
            sum += __shfl_xor(sum, 2, 32);
            sum += __shfl_xor(sum, 4, 32);
            sum += __shfl_xor(sum, 8, 32);
            l[r] = l[r] * fac[r] + sum;
            m[r] = mn;
        }
        if (lane == 0) {
#pragma unroll
            for (int r = 0; r < 8; ++r) fac_s[w][r] = fac[r];
        }
        if (lane == 16) {
#pragma unroll
            for (int r = 0; r < 8; ++r) fac_s[w][8 + r] = fac[r];
        }
        // P -> LDS (bf16) as [i][j] for the P^T B-fragment
#pragma unroll
        for (int t = 0; t < 4; ++t)
#pragma unroll
            for (int r = 0; r < 8; ++r)
                Ps[w][r + rowAdd][t * 16 + lh] = (__bf16)s[t][r];

        // rescale O by per-query factor (cols of O are queries)
        float fc = fac_s[w][lh];
#pragma unroll
        for (int dt = 0; dt < 4; ++dt)
#pragma unroll
            for (int e = 0; e < 8; ++e) Oacc[dt][e] *= fc;

        // O[d,i] += sum_j V[d,j] P[i,j]
#pragma unroll
        for (int dt = 0; dt < 4; ++dt)
#pragma unroll
            for (int kk = 0; kk < 2; ++kk) {
                v16bf av = fragA(&Vs[dt * 16 + lh][kk * 32], lane);
                v16bf bp = fragB(&Ps[w][lh][kk * 32], lane);
                Oacc[dt] = __builtin_amdgcn_wmma_f32_16x16x32_bf16(
                    false, av, false, bp, (short)0, Oacc[dt], false, false);
            }
    }

    if (lane == 0) {
#pragma unroll
        for (int r = 0; r < 8; ++r) lsum_s[w][r] = l[r];
    }
    if (lane == 16) {
#pragma unroll
        for (int r = 0; r < 8; ++r) lsum_s[w][8 + r] = l[r];
    }
    float linv = 1.0f / lsum_s[w][lh];
    int icol = iBase + w * 16 + lh;
#pragma unroll
    for (int dt = 0; dt < 4; ++dt)
#pragma unroll
        for (int r = 0; r < 8; ++r) {
            int d = dt * 16 + r + rowAdd;
            outp[(size_t)(b * 256 + h * 64 + d) * NPIX + icol] = Oacc[dt][r] * linv;
        }
}

// ---------------------------------------------------------------------------
// Depthwise 3x3 on v (+scale+bias), added to attention output; re-emit bf16.
__global__ __launch_bounds__(256) void pe_add_kernel(const __bf16* __restrict__ qkv,
                                                     const float* __restrict__ pe_w,
                                                     const float* __restrict__ pe_s,
                                                     const float* __restrict__ pe_b,
                                                     float* __restrict__ attnF,
                                                     __bf16* __restrict__ attnB)
{
    int idx = blockIdx.x;           // B*256 channels
    int b = idx >> 8, cc = idx & 255;
    int h = cc >> 6, d = cc & 63;
    const __bf16* v = qkv + (size_t)(b * 512 + h * 128 + 64 + d) * NPIX;
    float w9[9];
#pragma unroll
    for (int k = 0; k < 9; ++k) w9[k] = pe_w[cc * 9 + k];
    float sc = pe_s[cc], bi = pe_b[cc];
    float* oF = attnF + (size_t)(b * 256 + cc) * NPIX;
    __bf16* oB = attnB + (size_t)(b * 256 + cc) * NPIX;
    for (int p = threadIdx.x; p < NPIX; p += blockDim.x) {
        int y = p / 48, x = p - y * 48;
        float acc = 0.0f;
#pragma unroll
        for (int dy = -1; dy <= 1; ++dy)
#pragma unroll
            for (int dx = -1; dx <= 1; ++dx) {
                int yy = y + dy, xx = x + dx;
                if (yy >= 0 && yy < 48 && xx >= 0 && xx < 48)
                    acc += (float)v[yy * 48 + xx] * w9[(dy + 1) * 3 + (dx + 1)];
            }
        float r = oF[p] + acc * sc + bi;
        oF[p] = r;
        oB[p] = (__bf16)r;
    }
}

// ---------------------------------------------------------------------------
extern "C" void kernel_launch(void* const* d_in, const int* in_sizes, int n_in,
                              void* d_out, int out_size, void* d_ws, size_t ws_size,
                              hipStream_t stream)
{
    (void)in_sizes; (void)n_in; (void)out_size; (void)ws_size;
    const float* x      = (const float*)d_in[0];
    const float* cv1_w  = (const float*)d_in[1];
    const float* cv1_s  = (const float*)d_in[2];
    const float* cv1_b  = (const float*)d_in[3];
    const float* qkv_w  = (const float*)d_in[4];
    const float* qkv_s  = (const float*)d_in[5];
    const float* qkv_b  = (const float*)d_in[6];
    const float* pe_w   = (const float*)d_in[7];
    const float* pe_s   = (const float*)d_in[8];
    const float* pe_b   = (const float*)d_in[9];
    const float* proj_w = (const float*)d_in[10];
    const float* proj_s = (const float*)d_in[11];
    const float* proj_b = (const float*)d_in[12];
    const float* ffn1_w = (const float*)d_in[13];
    const float* ffn1_s = (const float*)d_in[14];
    const float* ffn1_b = (const float*)d_in[15];
    const float* ffn2_w = (const float*)d_in[16];
    const float* ffn2_s = (const float*)d_in[17];
    const float* ffn2_b = (const float*)d_in[18];
    const float* cv2_w  = (const float*)d_in[19];
    const float* cv2_s  = (const float*)d_in[20];
    const float* cv2_b  = (const float*)d_in[21];

    const size_t e512 = (size_t)BATCH * 512 * NPIX;   // 4.72M elems
    const size_t e256 = (size_t)BATCH * 256 * NPIX;

    char* base = (char*)d_ws;
    size_t off = 0;
    auto take = [&](size_t bytes) -> void* {
        void* p = base + off;
        off = (off + bytes + 255) & ~(size_t)255;
        return p;
    };
    __bf16* x_bf    = (__bf16*)take(e512 * 2);
    float*  ab_f32  = (float*) take(e512 * 4);
    __bf16* ab_bf   = (__bf16*)take(e512 * 2);
    __bf16* qkv_bf  = (__bf16*)take(e512 * 2);
    float*  attn_f  = (float*) take(e256 * 4);
    __bf16* attn_bf = (__bf16*)take(e256 * 2);
    float*  bb2_f   = (float*) take(e256 * 4);
    __bf16* bb2_bf  = (__bf16*)take(e256 * 2);
    __bf16* f_bf    = (__bf16*)take(e512 * 2);
    __bf16* bb3_bf  = (__bf16*)take(e256 * 2);

    dim3 blk(256);

    // 0) x -> bf16
    f32_to_bf16_kernel<<<dim3(2048), blk, 0, stream>>>(x, x_bf, e512);

    // 1) cv1 (512->512, SiLU): full fp32 (bb residual source) + bf16 (a & bb inputs)
    gemm_bf16_kernel<<<dim3(4, 18, BATCH), blk, 0, stream>>>(
        x_bf, 512, 0, x_bf, 512, 0, /*ksplit*/ 1 << 20,
        cv1_w, 512, 512, cv1_s, cv1_b, nullptr, 0, 0,
        ab_f32, 512, ab_bf, 512, /*act*/ 1);

    // 2) qkv (256->512, no act), input = bb (ab channels 256..511)
    gemm_bf16_kernel<<<dim3(4, 18, BATCH), blk, 0, stream>>>(
        ab_bf, 512, 256, ab_bf, 512, 256, 1 << 20,
        qkv_w, 256, 512, qkv_s, qkv_b, nullptr, 0, 0,
        nullptr, 0, qkv_bf, 512, 0);

    // 3) flash attention -> attn_f (pre position-encoding)
    flash_attn_kernel<<<dim3(18, 16), blk, 0, stream>>>(qkv_bf, attn_f, 0.17677669529663689f);

    // 4) + depthwise 3x3 of v
    pe_add_kernel<<<dim3(BATCH * 256), blk, 0, stream>>>(qkv_bf, pe_w, pe_s, pe_b, attn_f, attn_bf);

    // 5) proj (256->256) + residual bb -> bb2
    gemm_bf16_kernel<<<dim3(2, 18, BATCH), blk, 0, stream>>>(
        attn_bf, 256, 0, attn_bf, 256, 0, 1 << 20,
        proj_w, 256, 256, proj_s, proj_b, ab_f32, 512, 256,
        bb2_f, 256, bb2_bf, 256, 0);

    // 6) ffn1 (256->512, SiLU)
    gemm_bf16_kernel<<<dim3(4, 18, BATCH), blk, 0, stream>>>(
        bb2_bf, 256, 0, bb2_bf, 256, 0, 1 << 20,
        ffn1_w, 256, 512, ffn1_s, ffn1_b, nullptr, 0, 0,
        nullptr, 0, f_bf, 512, 1);

    // 7) ffn2 (512->256) + residual bb2 -> bb3
    gemm_bf16_kernel<<<dim3(2, 18, BATCH), blk, 0, stream>>>(
        f_bf, 512, 0, f_bf, 512, 0, 1 << 20,
        ffn2_w, 512, 256, ffn2_s, ffn2_b, bb2_f, 256, 0,
        nullptr, 0, bb3_bf, 256, 0);

    // 8) cv2 over concat(a, bb3): K split at 256, SiLU, fp32 output
    gemm_bf16_kernel<<<dim3(4, 18, BATCH), blk, 0, stream>>>(
        ab_bf, 512, 0, bb3_bf, 256, 0, /*ksplit*/ 256,
        cv2_w, 512, 512, cv2_s, cv2_b, nullptr, 0, 0,
        (float*)d_out, 512, nullptr, 0, 1);
}
